// ZXCalculusAttention_51642686767426
// MI455X (gfx1250) — matrix operand order repaired and checked by reading, not compiled, not run on hardware
//
#include <hip/hip_runtime.h>

typedef __bf16 bf16;
typedef __attribute__((ext_vector_type(16))) bf16 bf16x16;
typedef __attribute__((ext_vector_type(8)))  bf16 bf16x8;
typedef __attribute__((ext_vector_type(4)))  bf16 bf16x4;
typedef __attribute__((ext_vector_type(8)))  float f32x8;
typedef __attribute__((ext_vector_type(4)))  int   i32x4;

#define EDIM 2048
#define HN   16
#define DH   128
#define MTOT 16384   // B*S = 4*4096

// gfx1250 async global->LDS path (ASYNCcnt-tracked DMA writes into LDS).
#if __has_builtin(__builtin_amdgcn_global_load_async_to_lds_b128) && \
    __has_builtin(__builtin_amdgcn_s_wait_asynccnt)
#define HAVE_ASYNC 1
#else
#define HAVE_ASYNC 0
#endif

// builtin signature wants int4* in explicit address spaces:
//   arg0: global (AS1) int4*, arg1: LDS (AS3) int4*, then imm offset, imm cpol
#define GP(p) ((__attribute__((address_space(1))) i32x4*)(p))
#define LP(p) ((__attribute__((address_space(3))) i32x4*)(p))

// ---------------------------------------------------------------------------
// f32 -> bf16 conversion, 4 elements per thread (n divisible by 1024)
// ---------------------------------------------------------------------------
__global__ __launch_bounds__(256) void cvt_kernel(const float* __restrict__ src,
                                                  bf16* __restrict__ dst, int n) {
  int i = (blockIdx.x * 256 + threadIdx.x) * 4;
  if (i + 3 < n) {
    float4 v = *(const float4*)(src + i);
    bf16x4 o;
    o[0] = (bf16)v.x; o[1] = (bf16)v.y; o[2] = (bf16)v.z; o[3] = (bf16)v.w;
    *(bf16x4*)(dst + i) = o;
  }
}

// ---------------------------------------------------------------------------
// C[M,N] = A[M,K] * Bw[N,K]^T + bias   (nn.Linear NT layout)
// WG tile 128x256, BK=64, double-buffered LDS, async global->LDS staging.
// 8 waves in 2x4; wave tile 64x64 -> 32 WMMAs per K-tile (1 LDS read / WMMA).
// ---------------------------------------------------------------------------
#define BM 128
#define BN 256
#define BK 64
#define LDT 72   // padded LDS row stride (elems): 144B rows -> near-conflict-free

// stage one (BM x BK) A tile + (BN x BK) B tile into LDS
__device__ __forceinline__ void stage_tiles(const bf16* __restrict__ gA,
                                            const bf16* __restrict__ gB,
                                            bf16* sA, bf16* sB,
                                            int tid, int K) {
  int e = tid * 8;
#pragma unroll
  for (int r = 0; r < 4; ++r) {          // A: 128x64 = 8192 elems
    int row = e >> 6, col = e & 63;
#if HAVE_ASYNC
    __builtin_amdgcn_global_load_async_to_lds_b128(
        GP(gA + (size_t)row * K + col), LP(sA + row * LDT + col), 0, 0);
#else
    *(bf16x8*)(sA + row * LDT + col) = *(const bf16x8*)(gA + (size_t)row * K + col);
#endif
    e += 2048;
  }
  e = tid * 8;
#pragma unroll
  for (int r = 0; r < 8; ++r) {          // B: 256x64 = 16384 elems
    int row = e >> 6, col = e & 63;
#if HAVE_ASYNC
    __builtin_amdgcn_global_load_async_to_lds_b128(
        GP(gB + (size_t)row * K + col), LP(sB + row * LDT + col), 0, 0);
#else
    *(bf16x8*)(sB + row * LDT + col) = *(const bf16x8*)(gB + (size_t)row * K + col);
#endif
    e += 2048;
  }
}

template <int OUTF32>
__global__ __launch_bounds__(256) void gemm_bf16_nt(
    const bf16* __restrict__ A,    // [M,K]
    const bf16* __restrict__ Bw,   // [N,K]
    const float* __restrict__ bias,// [N]
    bf16* __restrict__ Cb, float* __restrict__ Cf,
    int M, int N, int K) {
  __shared__ bf16 As[2][BM * LDT];   // 2 x 18.4 KB
  __shared__ bf16 Bs[2][BN * LDT];   // 2 x 36.9 KB  -> 110.6 KB total

  const int tid  = threadIdx.x;
  const int wave = tid >> 5;
  const int lane = tid & 31;
  const int lh   = lane & 15;
  const int sel  = lane >> 4;     // which 16-lane half
  const int wm   = wave >> 2;     // 0..1 : 64-row block
  const int wn   = wave & 3;      // 0..3 : 64-col block
  const int blockM = blockIdx.y * BM;
  const int blockN = blockIdx.x * BN;

  const bf16* gA = A  + (size_t)blockM * K;  // tile base (advance by kt)
  const bf16* gB = Bw + (size_t)blockN * K;

  f32x8 acc[4][4] = {};

  int buf = 0;
  stage_tiles(gA, gB, As[0], Bs[0], tid, K);

  for (int kt = 0; kt < K; kt += BK) {
#if HAVE_ASYNC
    __builtin_amdgcn_s_wait_asynccnt(0);   // this buffer's DMA done
#endif
    __syncthreads();                        // visible to all waves

    if (kt + BK < K)
      stage_tiles(gA + kt + BK, gB + kt + BK, As[buf ^ 1], Bs[buf ^ 1], tid, K);

    // prefetch 2 tiles ahead into L2/WGP$ (one 128B row per prefetch)
    if (kt + 2 * BK < K) {
      if (tid < 128) {
        __builtin_prefetch(gA + (size_t)tid * K + kt + 2 * BK, 0, 3);
      } else {
        int pr = (tid - 128) * 2;
        __builtin_prefetch(gB + (size_t)pr * K + kt + 2 * BK, 0, 3);
        __builtin_prefetch(gB + (size_t)(pr + 1) * K + kt + 2 * BK, 0, 3);
      }
    }

    const bf16* sA = As[buf];
    const bf16* sB = Bs[buf];
#pragma unroll
    for (int ks = 0; ks < BK; ks += 32) {
      // A fragments: lane half sel -> K in {ks+sel*8..+7} U {ks+16+sel*8..+7}
      bf16x16 af[4];
#pragma unroll
      for (int mt = 0; mt < 4; ++mt) {
        const bf16* p = sA + (wm * 64 + mt * 16 + lh) * LDT + ks + sel * 8;
        bf16x8 lo = *(const bf16x8*)p;
        bf16x8 hi = *(const bf16x8*)(p + 16);
        af[mt] = __builtin_shufflevector(lo, hi, 0, 1, 2, 3, 4, 5, 6, 7,
                                         8, 9, 10, 11, 12, 13, 14, 15);
      }
      // B fragments: lane half sel -> K in {ks+sel*8..+7} U {ks+16+sel*8..+7}
      bf16x16 bfr[4];
#pragma unroll
      for (int nt = 0; nt < 4; ++nt) {
        const bf16* p = sB + (wn * 64 + nt * 16 + lh) * LDT + ks + sel * 8;
        bf16x8 lo = *(const bf16x8*)p;
        bf16x8 hi = *(const bf16x8*)(p + 16);
        bfr[nt] = __builtin_shufflevector(lo, hi, 0, 1, 2, 3, 4, 5, 6, 7,
                                          8, 9, 10, 11, 12, 13, 14, 15);
      }
#pragma unroll
      for (int mt = 0; mt < 4; ++mt)
#pragma unroll
        for (int nt = 0; nt < 4; ++nt)
          acc[mt][nt] = __builtin_amdgcn_wmma_f32_16x16x32_bf16(
              false, af[mt], false, bfr[nt], (short)0, acc[mt][nt], false, false);
    }
    buf ^= 1;
  }

  // epilogue: C element (m = base + sel*8 + j, n = base + lh)
#pragma unroll
  for (int nt = 0; nt < 4; ++nt) {
    int n = blockN + wn * 64 + nt * 16 + lh;
    float bv = bias[n];
#pragma unroll
    for (int mt = 0; mt < 4; ++mt) {
#pragma unroll
      for (int j = 0; j < 8; ++j) {
        int m = blockM + wm * 64 + mt * 16 + sel * 8 + j;
        float v = acc[mt][nt][j] + bv;
        if (OUTF32) Cf[(size_t)m * N + n] = v;
        else        Cb[(size_t)m * N + n] = (bf16)v;
      }
    }
  }
}

// ---------------------------------------------------------------------------
// Per-token head-mixing attention. One wave per token, 8 tokens per WG.
//   scores[h,g] = sum_d (z[h,d]*sin(phase[h,d])) * xs[g,d] / sqrt(Dh)
//   w = softmax_g(scores);  att[h,d] = sum_g w[h,g]*xs[g,d]
// ---------------------------------------------------------------------------
#define SPH_LD 132  // padded so lane h -> bank h*4 (no conflicts)

__global__ __launch_bounds__(256) void attn_kernel(
    const bf16* __restrict__ zb, const bf16* __restrict__ xsb,
    const float* __restrict__ phase, bf16* __restrict__ attb) {
  __shared__ float sph[HN * SPH_LD];
  __shared__ float lw[8 * 256];  // per-wave 16x16 softmax weights

  const int tid = threadIdx.x;
  for (int i = tid; i < HN * DH; i += 256) {
    int h = i >> 7, d = i & 127;
    sph[h * SPH_LD + d] = __sinf(phase[i]);
  }
  __syncthreads();

  const int wave = tid >> 5;
  const int lane = tid & 31;
  const int lh = lane & 15, sel = lane >> 4;
  const int t = blockIdx.x * 8 + wave;
  const bf16* zrow = zb  + (size_t)t * EDIM;
  const bf16* xrow = xsb + (size_t)t * EDIM;

  // ---- scores via 4 chained WMMAs over K = Dh = 128
  f32x8 c = {};
#pragma unroll
  for (int ks = 0; ks < 4; ++ks) {
    const int kb = ks * 32;
    bf16x8 zlo = *(const bf16x8*)(zrow + lh * DH + kb + sel * 8);
    bf16x8 zhi = *(const bf16x8*)(zrow + lh * DH + kb + sel * 8 + 16);
    const float* sp = sph + lh * SPH_LD + kb + sel * 8;
    bf16x16 a;
#pragma unroll
    for (int i = 0; i < 8; ++i) a[i]     = (bf16)((float)zlo[i] * sp[i]);
#pragma unroll
    for (int i = 0; i < 8; ++i) a[i + 8] = (bf16)((float)zhi[i] * sp[i + 16]);
    bf16x8 blo = *(const bf16x8*)(xrow + lh * DH + kb + sel * 8);
    bf16x8 bhi = *(const bf16x8*)(xrow + lh * DH + kb + sel * 8 + 16);
    bf16x16 b = __builtin_shufflevector(blo, bhi, 0, 1, 2, 3, 4, 5, 6, 7,
                                        8, 9, 10, 11, 12, 13, 14, 15);
    c = __builtin_amdgcn_wmma_f32_16x16x32_bf16(false, a, false, b, (short)0,
                                                c, false, false);
  }

  // ---- softmax over g: element (h = j + 8*sel, g = lh); reduce across
  //      the 16-lane half with xor shuffles (masks 1,2,4,8 stay in-half)
  const float scale = 0.08838834764831845f;  // 1/sqrt(128)
  float w[8];
#pragma unroll
  for (int j = 0; j < 8; ++j) {
    float s = c[j] * scale;
    float m = s;
    m = fmaxf(m, __shfl_xor(m, 1, 32));
    m = fmaxf(m, __shfl_xor(m, 2, 32));
    m = fmaxf(m, __shfl_xor(m, 4, 32));
    m = fmaxf(m, __shfl_xor(m, 8, 32));
    float ev = __expf(s - m);
    float su = ev;
    su += __shfl_xor(su, 1, 32);
    su += __shfl_xor(su, 2, 32);
    su += __shfl_xor(su, 4, 32);
    su += __shfl_xor(su, 8, 32);
    w[j] = ev / su;
  }
#pragma unroll
  for (int j = 0; j < 8; ++j)
    lw[wave * 256 + (j + 8 * sel) * 16 + lh] = w[j];
  __syncthreads();

  // ---- att = w @ xs_h : K=16 -> VALU with LDS-broadcast weights
  const int d0 = lane * 4;  // lane owns 4 consecutive d
  float4 xv[16];
#pragma unroll
  for (int g = 0; g < 16; ++g) {
    bf16x4 xt = *(const bf16x4*)(xrow + g * DH + d0);
    xv[g] = make_float4((float)xt[0], (float)xt[1], (float)xt[2], (float)xt[3]);
  }
  const float* wl = lw + wave * 256;
  bf16* arow = attb + (size_t)t * EDIM;
  for (int hh = 0; hh < HN; ++hh) {
    float a0 = 0.f, a1 = 0.f, a2 = 0.f, a3 = 0.f;
#pragma unroll
    for (int g = 0; g < 16; ++g) {
      float wv = wl[hh * 16 + g];
      a0 += wv * xv[g].x; a1 += wv * xv[g].y;
      a2 += wv * xv[g].z; a3 += wv * xv[g].w;
    }
    bf16x4 o;
    o[0] = (bf16)a0; o[1] = (bf16)a1; o[2] = (bf16)a2; o[3] = (bf16)a3;
    *(bf16x4*)(arow + hh * DH + d0) = o;
  }
}

// ---------------------------------------------------------------------------
extern "C" void kernel_launch(void* const* d_in, const int* in_sizes, int n_in,
                              void* d_out, int out_size, void* d_ws, size_t ws_size,
                              hipStream_t stream) {
  const float* x     = (const float*)d_in[0];
  const float* Wz    = (const float*)d_in[1];
  const float* bz    = (const float*)d_in[2];
  const float* Wx    = (const float*)d_in[3];
  const float* bx    = (const float*)d_in[4];
  const float* phase = (const float*)d_in[5];
  const float* Wo    = (const float*)d_in[6];
  const float* bo    = (const float*)d_in[7];
  float* out = (float*)d_out;

  const size_t SZ_X = (size_t)MTOT * EDIM * sizeof(bf16);   // 67 MB
  const size_t SZ_W = (size_t)EDIM * EDIM * sizeof(bf16);   // 8.4 MB

  char* ws = (char*)d_ws;
  bf16* xb  = (bf16*)(ws);
  bf16* Wzb = (bf16*)(ws + SZ_X);
  bf16* Wxb = (bf16*)(ws + SZ_X + SZ_W);
  bf16* Wob = (bf16*)(ws + SZ_X + 2 * SZ_W);
  bf16* zb  = (bf16*)(ws + SZ_X + 3 * SZ_W);
  bf16* xsb = (bf16*)(ws + 2 * SZ_X + 3 * SZ_W);
  bf16* attb = xb;  // x no longer needed after the first two GEMMs

  const int NX = MTOT * EDIM;   // 33,554,432
  const int NW = EDIM * EDIM;   //  4,194,304
  cvt_kernel<<<NX / 1024, 256, 0, stream>>>(x, xb, NX);
  cvt_kernel<<<NW / 1024, 256, 0, stream>>>(Wz, Wzb, NW);
  cvt_kernel<<<NW / 1024, 256, 0, stream>>>(Wx, Wxb, NW);
  cvt_kernel<<<NW / 1024, 256, 0, stream>>>(Wo, Wob, NW);

  dim3 gg(EDIM / BN, MTOT / BM);  // (8, 128)
  gemm_bf16_nt<0><<<gg, 256, 0, stream>>>(xb, Wzb, bz, zb, nullptr,
                                          MTOT, EDIM, EDIM);
  gemm_bf16_nt<0><<<gg, 256, 0, stream>>>(xb, Wxb, bx, xsb, nullptr,
                                          MTOT, EDIM, EDIM);

  attn_kernel<<<MTOT / 8, 256, 0, stream>>>(zb, xsb, phase, attb);

  gemm_bf16_nt<1><<<gg, 256, 0, stream>>>(attb, Wob, bo, nullptr, out,
                                          MTOT, EDIM, EDIM);
}
